// MSAColumnAttention_81338090651655
// MI455X (gfx1250) — matrix operand order repaired and checked by reading, not compiled, not run on hardware
//
#include <hip/hip_runtime.h>
#include <cstdint>

// ---------------- problem constants ----------------
#define NSEQ 256
#define NRES 384
#define CM   256
#define NH   8
#define CH   32
#define NROW (NSEQ * NRES) /* 98304 flattened (r,s) rows */

// ---------------- vector types (trivially-constructible, union-safe) ----------
typedef __attribute__((ext_vector_type(16))) __bf16       v16bf;
typedef __attribute__((ext_vector_type(8)))  float        v8f;
typedef __attribute__((ext_vector_type(4)))  float        f32x4;
typedef __attribute__((ext_vector_type(4)))  unsigned int u32x4;
typedef __attribute__((ext_vector_type(8)))  int          i32x8;
typedef __attribute__((ext_vector_type(4)))  int          i32x4;

union Frag {
  v16bf v;
  u32x4 q[2];
};

// ---------------- helpers ----------------
__device__ __forceinline__ unsigned short f32_to_bf16(float f) {
  unsigned int u = __float_as_uint(f);
  unsigned int r = 0x7FFFu + ((u >> 16) & 1u); // round-to-nearest-even
  return (unsigned short)((u + r) >> 16);
}
__device__ __forceinline__ unsigned int pack_bf16x2(float a, float b) {
  return (unsigned int)f32_to_bf16(a) | ((unsigned int)f32_to_bf16(b) << 16);
}
// reductions across the 16-lane half-wave (WMMA C-layout rows live in lanes 0-15 / 16-31)
__device__ __forceinline__ float rmax16(float x) {
  x = fmaxf(x, __shfl_xor(x, 1, 32));
  x = fmaxf(x, __shfl_xor(x, 2, 32));
  x = fmaxf(x, __shfl_xor(x, 4, 32));
  x = fmaxf(x, __shfl_xor(x, 8, 32));
  return x;
}
__device__ __forceinline__ float rsum16(float x) {
  x += __shfl_xor(x, 1, 32);
  x += __shfl_xor(x, 2, 32);
  x += __shfl_xor(x, 4, 32);
  x += __shfl_xor(x, 8, 32);
  return x;
}
__device__ __forceinline__ float rsum32(float x) {
  x = rsum16(x);
  x += __shfl_xor(x, 16, 32);
  return x;
}

// ---------------------------------------------------------------------
// Tensor Data Mover: issue a 1-row D# that DMAs `nelems` bf16 elements
// (contiguous) from global memory into LDS at byte offset lds_off.
// D# bit layout per CDNA5 ISA 8.3/8.4:
//   group0: count=1 | lds_addr | global_addr[56:0] | type=2
//   group1: data_size=1(2B), tensor_dim0=tile_dim0=nelems,
//           tensor_dim1=tile_dim1=1, tensor_dim0_stride=nelems
//   group2/3: zero (<=2D tensor)
// This toolchain exposes the 6-arg builtin (extra i32x8 group, zeroed).
// Caller: exactly one wave should issue; drain with s_wait_tensorcnt(0).
// ---------------------------------------------------------------------
__device__ __forceinline__ void tdm_load_bf16_1d(const void* gsrc,
                                                 unsigned lds_off,
                                                 unsigned nelems /* <= 65535 */) {
  unsigned long long ga = (unsigned long long)(size_t)gsrc;
  u32x4 g0;
  g0[0] = 1u;                                        // count=1, user descriptor
  g0[1] = lds_off;                                   // lds_addr (bytes)
  g0[2] = (unsigned)(ga & 0xFFFFFFFFu);              // global_addr[31:0]
  g0[3] = (unsigned)((ga >> 32) & 0x01FFFFFFu) |     // global_addr[56:32]
          (2u << 30);                                // type=2 ("image")
  i32x8 g1;
  g1[0] = (int)(1u << 16);                           // data_size=1 -> 2 bytes
  g1[1] = (int)(nelems << 16);                       // tensor_dim0[15:0] @ bits 63:48
  g1[2] = (int)((nelems >> 16) | (1u << 16));        // tensor_dim0[31:16] | tensor_dim1=1
  g1[3] = (int)(nelems << 16);                       // tile_dim0 @ bits 127:112
  g1[4] = 1;                                         // tile_dim1=1 (tile_dim2=0)
  g1[5] = (int)nelems;                               // tensor_dim0_stride[31:0]
  g1[6] = 0;                                         // stride hi / dim1_stride lo
  g1[7] = 0;
  i32x4 gz4 = {0, 0, 0, 0};
  i32x8 gz8 = {0, 0, 0, 0, 0, 0, 0, 0};
  __builtin_amdgcn_tensor_load_to_lds(g0, g1, gz4, gz4, gz8, 0);
}

// =======================================================================
// Kernel 0: transpose+convert 5 weight matrices (256x256 f32 [k][n]) into
// bf16 W^T [n][k] so B-fragments are 16-contiguous-k loads.
// =======================================================================
__global__ __launch_bounds__(256) void k_prep_weights(
    const float* __restrict__ Wq, const float* __restrict__ Wk,
    const float* __restrict__ Wv, const float* __restrict__ Wg,
    const float* __restrict__ Wo, unsigned short* __restrict__ wts) {
  int mat = blockIdx.y;
  const float* W = (mat == 0) ? Wq : (mat == 1) ? Wk : (mat == 2) ? Wv
                 : (mat == 3) ? Wg : Wo;
  int e = blockIdx.x * 256 + threadIdx.x; // 0..65535
  int k = e >> 8, n = e & 255;
  wts[(size_t)mat * 65536 + (size_t)n * 256 + k] = f32_to_bf16(W[e]);
}

// =======================================================================
// Kernel 1: LayerNorm over c_m, transpose (S,R,C)->(R,S,C), cast bf16.
// One wave per row; wave32 shuffle reduction; 32B vector loads, 16B store.
// =======================================================================
__global__ __launch_bounds__(256) void k_layernorm(
    const float* __restrict__ m, const float* __restrict__ lng,
    const float* __restrict__ lnb, unsigned short* __restrict__ xb) {
  int wave = threadIdx.x >> 5, lane = threadIdx.x & 31;
  int row = blockIdx.x * 8 + wave; // row = s*NRES + r
  int s = row / NRES;
  int r = row - s * NRES;
  const float* src = m + (size_t)row * CM + lane * 8;
  f32x4 a0 = *(const f32x4*)src;
  f32x4 a1 = *(const f32x4*)(src + 4);
  float sum = 0.f, sq = 0.f;
#pragma unroll
  for (int i = 0; i < 4; ++i) { sum += a0[i]; sq += a0[i] * a0[i]; }
#pragma unroll
  for (int i = 0; i < 4; ++i) { sum += a1[i]; sq += a1[i] * a1[i]; }
  sum = rsum32(sum);
  sq  = rsum32(sq);
  float mu   = sum * (1.0f / CM);
  float var  = sq * (1.0f / CM) - mu * mu;
  float rstd = rsqrtf(var + 1e-5f);
  const float* gp = lng + lane * 8;
  const float* bp = lnb + lane * 8;
  f32x4 g0 = *(const f32x4*)gp, g1 = *(const f32x4*)(gp + 4);
  f32x4 b0 = *(const f32x4*)bp, b1 = *(const f32x4*)(bp + 4);
  float o[8];
#pragma unroll
  for (int i = 0; i < 4; ++i) o[i]     = (a0[i] - mu) * rstd * g0[i] + b0[i];
#pragma unroll
  for (int i = 0; i < 4; ++i) o[i + 4] = (a1[i] - mu) * rstd * g1[i] + b1[i];
  u32x4 pk;
  pk[0] = pack_bf16x2(o[0], o[1]);
  pk[1] = pack_bf16x2(o[2], o[3]);
  pk[2] = pack_bf16x2(o[4], o[5]);
  pk[3] = pack_bf16x2(o[6], o[7]);
  *(u32x4*)(xb + ((size_t)(r * NSEQ + s)) * CM + lane * 8) = pk;
}

// =======================================================================
// Kernel 2: projections y = x @ W for W in {Wq,Wk,Wv,Wg}.
// M=98304, N=256 (split: blockIdx.z half), K=256. 8 waves/block, wave tile
// 16x128, K in 8 steps of 32 -> one v_wmma_f32_16x16x32_bf16 per 16x16 tile.
// W^T half (64KB) staged in LDS by the TDM. q,k -> [r][h][s][c];
// v -> transposed [r][h][c][s]; g -> sigmoid(y+bg) in f32.
// =======================================================================
__global__ __launch_bounds__(256) void k_proj(
    const unsigned short* __restrict__ xb, const unsigned short* __restrict__ wts,
    const float* __restrict__ bg,
    unsigned short* __restrict__ qb, unsigned short* __restrict__ kb,
    unsigned short* __restrict__ vtb, float* __restrict__ gbuf) {
  extern __shared__ unsigned short lds[]; // 32768 ushorts = 64 KB
  int widx  = blockIdx.y;                 // 0=q 1=k 2=v 3=g
  int nhalf = blockIdx.z;                 // n in [nhalf*128, nhalf*128+128)
  const unsigned short* W = wts + (size_t)widx * 65536 + (size_t)nhalf * 128 * 256;
  if (threadIdx.x < 32) { // wave 0 drives the DMA engine
    tdm_load_bf16_1d(W, (unsigned)(size_t)lds, 32768);
    __builtin_amdgcn_s_wait_tensorcnt(0);
  }
  __syncthreads();

  int wave = threadIdx.x >> 5, lane = threadIdx.x & 31;
  int lan15 = lane & 15, half = lane >> 4;
  int m0 = blockIdx.x * 128 + wave * 16;
  const unsigned short* arow = xb + (size_t)(m0 + lan15) * CM;

  const v8f vzero = {0.f, 0.f, 0.f, 0.f, 0.f, 0.f, 0.f, 0.f};
  v8f acc[8];
#pragma unroll
  for (int nt = 0; nt < 8; ++nt) acc[nt] = vzero;

#pragma unroll
  for (int kc = 0; kc < 8; ++kc) {
    Frag A;
    A.q[0] = *(const u32x4*)(arow + kc * 32 + half * 16);
    A.q[1] = *(const u32x4*)(arow + kc * 32 + half * 16 + 8);
    if (kc < 7) __builtin_prefetch(arow + (kc + 1) * 32, 0, 1); // global_prefetch
#pragma unroll
    for (int nt = 0; nt < 8; ++nt) {
      Frag B;
      const unsigned short* bp = lds + (size_t)(nt * 16 + lan15) * 256 + kc * 32 + half * 16;
      B.q[0] = *(const u32x4*)bp;
      B.q[1] = *(const u32x4*)(bp + 8);
      acc[nt] = __builtin_amdgcn_wmma_f32_16x16x32_bf16(
          false, A.v, false, B.v, (short)0, acc[nt], false, false);
    }
  }

  int r = m0 >> 8; // 128-row tiles never straddle an r boundary (256 | tiles)
#pragma unroll
  for (int nt = 0; nt < 8; ++nt) {
    int n = nhalf * 128 + nt * 16 + lan15;
    int h = n >> 5, c = n & 31;
#pragma unroll
    for (int j = 0; j < 8; ++j) {
      int row = m0 + half * 8 + j;
      int s = row & 255;
      float val = acc[nt][j];
      if (widx == 3) {
        gbuf[(size_t)row * 256 + n] = 1.0f / (1.0f + __expf(-(val + bg[n])));
      } else {
        unsigned short bv = f32_to_bf16(val);
        if (widx == 2)
          vtb[(size_t)((r * NH + h) * CH + c) * NSEQ + s] = bv; // V transposed
        else {
          unsigned short* dst = (widx == 0) ? qb : kb;
          dst[(size_t)((r * NH + h) * NSEQ + s) * CH + c] = bv;
        }
      }
    }
  }
}

// =======================================================================
// Kernel 3: attention per (r,h). Block = 128 threads (4 waves).
// K tile (256x32) and V^T tile (32x256) staged in LDS by the TDM. Each wave
// does four 16-row tiles: scores (16 WMMA, K=c=32 -> one WMMA per tile),
// f32 softmax with lane-group-16 reductions, probs -> bf16 via per-wave LDS
// buffer (C-layout -> A-layout), P@V (16 WMMA), fused 1/rowsum and sigmoid
// gate. LDS: 16KB K + 16KB V^T + 32KB probs = 64KB.
// =======================================================================
__global__ __launch_bounds__(128) void k_attn(
    const unsigned short* __restrict__ qb, const unsigned short* __restrict__ kb,
    const unsigned short* __restrict__ vtb, const float* __restrict__ gbuf,
    unsigned short* __restrict__ oatt) {
  extern __shared__ unsigned short lds[];
  unsigned short* Klds = lds;          // 8192
  unsigned short* Vlds = lds + 8192;   // 8192
  unsigned short* Plds = lds + 16384;  // 16384 (4 waves * 4096)

  int rh = blockIdx.x;
  int r = rh >> 3, h = rh & 7;
  const unsigned short* Kg = kb + (size_t)rh * NSEQ * CH;
  const unsigned short* Vg = vtb + (size_t)rh * NSEQ * CH;
  if (threadIdx.x < 32) { // wave 0 issues both tile DMAs, drains TENSORcnt
    tdm_load_bf16_1d(Kg, (unsigned)(size_t)Klds, 8192);
    tdm_load_bf16_1d(Vg, (unsigned)(size_t)Vlds, 8192);
    __builtin_amdgcn_s_wait_tensorcnt(0);
  }
  __syncthreads();

  int wave = threadIdx.x >> 5, lane = threadIdx.x & 31;
  int lan15 = lane & 15, half = lane >> 4;
  const unsigned short* Qg = qb + (size_t)rh * NSEQ * CH;
  unsigned short* Pw = Plds + wave * 4096;
  const float SCL = 0.17677669529663687f * 1.4426950408889634f; // 1/sqrt(32)*log2(e)
  const v8f vzero = {0.f, 0.f, 0.f, 0.f, 0.f, 0.f, 0.f, 0.f};

  for (int rt = 0; rt < 4; ++rt) {
    int m0 = wave * 64 + rt * 16;
    Frag A;
    const unsigned short* qrow = Qg + (size_t)(m0 + lan15) * CH + half * 16;
    A.q[0] = *(const u32x4*)qrow;
    A.q[1] = *(const u32x4*)(qrow + 8);

    v8f sc[16];
#pragma unroll
    for (int jt = 0; jt < 16; ++jt) {
      Frag B;
      const unsigned short* bp = Klds + (size_t)(jt * 16 + lan15) * CH + half * 16;
      B.q[0] = *(const u32x4*)bp;
      B.q[1] = *(const u32x4*)(bp + 8);
      sc[jt] = __builtin_amdgcn_wmma_f32_16x16x32_bf16(
          false, A.v, false, B.v, (short)0, vzero, false, false);
    }

    float rsum[8];
#pragma unroll
    for (int j = 0; j < 8; ++j) {
      float mx = -3.0e38f;
#pragma unroll
      for (int jt = 0; jt < 16; ++jt) mx = fmaxf(mx, sc[jt][j]);
      mx = rmax16(mx);
      float sum = 0.f;
#pragma unroll
      for (int jt = 0; jt < 16; ++jt) {
        float p = exp2f((sc[jt][j] - mx) * SCL);
        sc[jt][j] = p;
        sum += p;
      }
      rsum[j] = rsum16(sum);
    }

    // C-layout -> A-layout via per-wave LDS buffer (bf16 probs)
#pragma unroll
    for (int jt = 0; jt < 16; ++jt)
#pragma unroll
      for (int j = 0; j < 8; ++j)
        Pw[(size_t)(half * 8 + j) * 256 + jt * 16 + lan15] = f32_to_bf16(sc[jt][j]);

    v8f o0 = vzero, o1 = vzero;
#pragma unroll
    for (int kc = 0; kc < 8; ++kc) {
      Frag P, B0, B1;
      const unsigned short* pr = Pw + (size_t)lan15 * 256 + kc * 32 + half * 16;
      P.q[0] = *(const u32x4*)pr;
      P.q[1] = *(const u32x4*)(pr + 8);
      const unsigned short* v0p = Vlds + (size_t)lan15 * 256 + kc * 32 + half * 16;
      const unsigned short* v1p = Vlds + (size_t)(16 + lan15) * 256 + kc * 32 + half * 16;
      B0.q[0] = *(const u32x4*)v0p; B0.q[1] = *(const u32x4*)(v0p + 8);
      B1.q[0] = *(const u32x4*)v1p; B1.q[1] = *(const u32x4*)(v1p + 8);
      o0 = __builtin_amdgcn_wmma_f32_16x16x32_bf16(false, P.v, false, B0.v, (short)0, o0, false, false);
      o1 = __builtin_amdgcn_wmma_f32_16x16x32_bf16(false, P.v, false, B1.v, (short)0, o1, false, false);
    }

#pragma unroll
    for (int j = 0; j < 8; ++j) {
      int srow = m0 + half * 8 + j;
      float inv = 1.0f / rsum[j];
      size_t base = ((size_t)(r * NSEQ + srow)) * (NH * CH) + h * CH;
      float g0 = gbuf[base + lan15];
      float g1 = gbuf[base + 16 + lan15];
      oatt[base + lan15]      = f32_to_bf16(o0[j] * inv * g0);
      oatt[base + 16 + lan15] = f32_to_bf16(o1[j] * inv * g1);
    }
  }
}

// =======================================================================
// Kernel 4: out = o' @ Wo + bo, scattered back to (S,R,C) f32.
// Same GEMM tiling as k_proj; Wo^T half staged in LDS by the TDM.
// =======================================================================
__global__ __launch_bounds__(256) void k_outproj(
    const unsigned short* __restrict__ oatt, const unsigned short* __restrict__ wts,
    const float* __restrict__ bo, float* __restrict__ out) {
  extern __shared__ unsigned short lds[];
  int nhalf = blockIdx.y;
  const unsigned short* W = wts + (size_t)4 * 65536 + (size_t)nhalf * 128 * 256;
  if (threadIdx.x < 32) {
    tdm_load_bf16_1d(W, (unsigned)(size_t)lds, 32768);
    __builtin_amdgcn_s_wait_tensorcnt(0);
  }
  __syncthreads();

  int wave = threadIdx.x >> 5, lane = threadIdx.x & 31;
  int lan15 = lane & 15, half = lane >> 4;
  int m0 = blockIdx.x * 128 + wave * 16;
  const unsigned short* arow = oatt + (size_t)(m0 + lan15) * CM;

  const v8f vzero = {0.f, 0.f, 0.f, 0.f, 0.f, 0.f, 0.f, 0.f};
  v8f acc[8];
#pragma unroll
  for (int nt = 0; nt < 8; ++nt) acc[nt] = vzero;

#pragma unroll
  for (int kc = 0; kc < 8; ++kc) {
    Frag A;
    A.q[0] = *(const u32x4*)(arow + kc * 32 + half * 16);
    A.q[1] = *(const u32x4*)(arow + kc * 32 + half * 16 + 8);
    if (kc < 7) __builtin_prefetch(arow + (kc + 1) * 32, 0, 1);
#pragma unroll
    for (int nt = 0; nt < 8; ++nt) {
      Frag B;
      const unsigned short* bp = lds + (size_t)(nt * 16 + lan15) * 256 + kc * 32 + half * 16;
      B.q[0] = *(const u32x4*)bp;
      B.q[1] = *(const u32x4*)(bp + 8);
      acc[nt] = __builtin_amdgcn_wmma_f32_16x16x32_bf16(
          false, A.v, false, B.v, (short)0, acc[nt], false, false);
    }
  }

#pragma unroll
  for (int nt = 0; nt < 8; ++nt) {
    int n = nhalf * 128 + nt * 16 + lan15;
    float bias = bo[n];
#pragma unroll
    for (int j = 0; j < 8; ++j) {
      int row = m0 + half * 8 + j; // row = r*256 + s
      int rr = row >> 8, ss = row & 255;
      out[((size_t)ss * NRES + rr) * CM + n] = acc[nt][j] + bias;
    }
  }
}

// =======================================================================
// Host-side launch. Workspace layout (~353 MB):
//   xb   bf16  [R][S][C]        50331648 B
//   q    bf16  [r][h][s][c]     50331648 B
//   k    bf16  [r][h][s][c]     50331648 B
//   vT   bf16  [r][h][c][s]     50331648 B
//   g    f32   [r][s][hc]      100663296 B
//   o'   bf16  [r][s][hc]       50331648 B
//   W^T  bf16  5 x 256x256        655360 B
// =======================================================================
extern "C" void kernel_launch(void* const* d_in, const int* in_sizes, int n_in,
                              void* d_out, int out_size, void* d_ws, size_t ws_size,
                              hipStream_t stream) {
  (void)in_sizes; (void)n_in; (void)out_size; (void)ws_size;
  const float* m    = (const float*)d_in[0];
  const float* ln_g = (const float*)d_in[1];
  const float* ln_b = (const float*)d_in[2];
  const float* Wq   = (const float*)d_in[3];
  const float* Wk   = (const float*)d_in[4];
  const float* Wv   = (const float*)d_in[5];
  const float* Wg   = (const float*)d_in[6];
  const float* bg   = (const float*)d_in[7];
  const float* Wo   = (const float*)d_in[8];
  const float* bo   = (const float*)d_in[9];

  char* ws = (char*)d_ws;
  const size_t NB   = (size_t)NROW * CM; // 25165824 elements
  unsigned short* xb   = (unsigned short*)(ws);
  unsigned short* qb   = (unsigned short*)(ws + NB * 2);
  unsigned short* kbuf = (unsigned short*)(ws + NB * 4);
  unsigned short* vtb  = (unsigned short*)(ws + NB * 6);
  float*          gbuf = (float*)         (ws + NB * 8);
  unsigned short* oatt = (unsigned short*)(ws + NB * 12);
  unsigned short* wts  = (unsigned short*)(ws + NB * 14);

  k_prep_weights<<<dim3(256, 5), 256, 0, stream>>>(Wq, Wk, Wv, Wg, Wo, wts);
  k_layernorm<<<NROW / 8, 256, 0, stream>>>(m, ln_g, ln_b, xb);
  k_proj<<<dim3(NROW / 128, 4, 2), 256, 65536, stream>>>(xb, wts, bg, qb, kbuf, vtb, gbuf);
  k_attn<<<NRES * NH, 128, 65536, stream>>>(qb, kbuf, vtb, gbuf, oatt);
  k_outproj<<<dim3(NROW / 128, 2), 256, 65536, stream>>>(oatt, wts, bo, (float*)d_out);
}
